// TripartiteHeteroGNN__7198365188424
// MI455X (gfx1250) — compile-verified, matrix operand order checked by reading, not run on hardware
//
#include <hip/hip_runtime.h>
#include <hip/hip_bf16.h>

// ---------------------------------------------------------------------------
// Types for CDNA5 WMMA (wave32): v_wmma_f32_16x16x32_bf16
// ---------------------------------------------------------------------------
typedef __bf16 bf16_t;
typedef __attribute__((ext_vector_type(16))) bf16_t          v16bf;
typedef __attribute__((ext_vector_type(8)))  float           v8f;
typedef __attribute__((ext_vector_type(16))) unsigned short  u16v;

#define HID   64
#define HID2  128

__device__ __forceinline__ unsigned short bf16r(float f) {
  unsigned u = __builtin_bit_cast(unsigned, f);
  unsigned r = (u + 0x7FFFu + ((u >> 16) & 1u)) >> 16;   // round-to-nearest-even
  return (unsigned short)r;
}

__device__ __forceinline__ v8f v8f_zero() {
  v8f z;
#pragma unroll
  for (int i = 0; i < 8; ++i) z[i] = 0.f;
  return z;
}

__device__ __forceinline__ v16bf v16bf_zero() {
  u16v t;
#pragma unroll
  for (int i = 0; i < 16; ++i) t[i] = 0;
  return __builtin_bit_cast(v16bf, t);
}

__device__ __forceinline__ v8f wmma_bf16(v16bf a, v16bf b, v8f c) {
  // D(f32 16x16) = A(bf16 16x32) * B(bf16 32x16) + C
  return __builtin_amdgcn_wmma_f32_16x16x32_bf16(false, a, false, b, (short)0, c,
                                                 false, false);
}

// A-operand (16-bit, 16x32) lane layout: lane l holds row m = l%16; element j
// maps to K = (j/8)*16 + (l/16)*8 + (j%8)  ->  two contiguous runs of 8.
__device__ __forceinline__ v16bf load_a_sum(const float* __restrict__ p0,
                                            const float* __restrict__ p1,
                                            int kbase, int g) {
  u16v t;
#pragma unroll
  for (int hh = 0; hh < 2; ++hh) {
    const int off = kbase + hh * 16 + g * 8;
    const float4 a0 = *(const float4*)(p0 + off);
    const float4 a1 = *(const float4*)(p0 + off + 4);
    const float4 b0 = *(const float4*)(p1 + off);
    const float4 b1 = *(const float4*)(p1 + off + 4);
    t[hh * 8 + 0] = bf16r(a0.x + b0.x);
    t[hh * 8 + 1] = bf16r(a0.y + b0.y);
    t[hh * 8 + 2] = bf16r(a0.z + b0.z);
    t[hh * 8 + 3] = bf16r(a0.w + b0.w);
    t[hh * 8 + 4] = bf16r(a1.x + b1.x);
    t[hh * 8 + 5] = bf16r(a1.y + b1.y);
    t[hh * 8 + 6] = bf16r(a1.z + b1.z);
    t[hh * 8 + 7] = bf16r(a1.w + b1.w);
  }
  return __builtin_bit_cast(v16bf, t);
}

__device__ __forceinline__ v16bf load_a_one(const float* __restrict__ p0,
                                            int kbase, int g) {
  u16v t;
#pragma unroll
  for (int hh = 0; hh < 2; ++hh) {
    const int off = kbase + hh * 16 + g * 8;
    const float4 a0 = *(const float4*)(p0 + off);
    const float4 a1 = *(const float4*)(p0 + off + 4);
    t[hh * 8 + 0] = bf16r(a0.x);
    t[hh * 8 + 1] = bf16r(a0.y);
    t[hh * 8 + 2] = bf16r(a0.z);
    t[hh * 8 + 3] = bf16r(a0.w);
    t[hh * 8 + 4] = bf16r(a1.x);
    t[hh * 8 + 5] = bf16r(a1.y);
    t[hh * 8 + 6] = bf16r(a1.z);
    t[hh * 8 + 7] = bf16r(a1.w);
  }
  return __builtin_bit_cast(v16bf, t);
}

// A-operand from LDS f32 row (post-GEMM1 activations)
__device__ __forceinline__ v16bf load_a_lds(const float* row, int kbase, int g) {
  u16v t;
#pragma unroll
  for (int hh = 0; hh < 2; ++hh) {
    const int off = kbase + hh * 16 + g * 8;
#pragma unroll
    for (int i = 0; i < 8; ++i) t[hh * 8 + i] = bf16r(row[off + i]);
  }
  return __builtin_bit_cast(v16bf, t);
}

// A-operand with fused batch-norm (scale/shift) + relu
__device__ __forceinline__ v16bf load_a_bn(const float* __restrict__ hrow,
                                           const float* __restrict__ ss,
                                           int kbase, int g) {
  u16v t;
#pragma unroll
  for (int hh = 0; hh < 2; ++hh) {
    const int off = kbase + hh * 16 + g * 8;
#pragma unroll
    for (int i = 0; i < 8; ++i) {
      float v = fmaf(hrow[off + i], ss[off + i], ss[64 + off + i]);
      t[hh * 8 + i] = bf16r(fmaxf(v, 0.f));
    }
  }
  return __builtin_bit_cast(v16bf, t);
}

// ---------------------------------------------------------------------------
// Utility kernels
// ---------------------------------------------------------------------------
__global__ void zero_kernel(float* __restrict__ p, int count) {
  int i = blockIdx.x * blockDim.x + threadIdx.x;
  if (i < count) p[i] = 0.f;
}

// Pre-swizzle a row-major f32 weight [K x Ncol] into per-lane WMMA B-operand
// layout: tile t = kt*(Ncol/16)+nt, then 32 lanes x 16 bf16, fully contiguous.
__global__ void pack_b_kernel(const float* __restrict__ W,
                              unsigned short* __restrict__ out,
                              int K, int Ncol) {
  int idx = blockIdx.x * blockDim.x + threadIdx.x;
  int total = (K >> 5) * (Ncol >> 4) * 512;
  if (idx >= total) return;
  int t    = idx >> 9;
  int lane = (idx >> 4) & 31;
  int j    = idx & 15;
  int NT = Ncol >> 4;
  int kt = t / NT, nt = t % NT;
  int g = lane >> 4, n = lane & 15;
  int kk = ((j >> 3) << 4) + g * 8 + (j & 7);
  out[idx] = bf16r(W[(kt * 32 + kk) * Ncol + nt * 16 + n]);
}

// ---------------------------------------------------------------------------
// Encoder: h = x@W1+b1  (x is [N,2])
// ---------------------------------------------------------------------------
__global__ void enc_h_kernel(const float* __restrict__ x,
                             const float* __restrict__ W1,
                             const float* __restrict__ b1,
                             float* __restrict__ h, int N) {
  int idx = blockIdx.x * blockDim.x + threadIdx.x;
  if (idx >= N * HID) return;
  int n = idx >> 6, j = idx & 63;
  h[idx] = fmaf(x[n * 2], W1[j], fmaf(x[n * 2 + 1], W1[64 + j], b1[j]));
}

// Column sums / sums-of-squares (one block per column)
__global__ __launch_bounds__(256) void enc_stats_kernel(
    const float* __restrict__ h, float* __restrict__ stats, int N) {
  __shared__ float rs[256], rq[256];
  int j = blockIdx.x;
  float s = 0.f, q = 0.f;
  for (int n = threadIdx.x; n < N; n += 256) {
    float v = h[n * HID + j];
    s += v; q += v * v;
  }
  rs[threadIdx.x] = s; rq[threadIdx.x] = q;
  __syncthreads();
  for (int d = 128; d > 0; d >>= 1) {
    if (threadIdx.x < d) {
      rs[threadIdx.x] += rs[threadIdx.x + d];
      rq[threadIdx.x] += rq[threadIdx.x + d];
    }
    __syncthreads();
  }
  if (threadIdx.x == 0) { stats[j] = rs[0]; stats[64 + j] = rq[0]; }
}

// Fold BN into per-column scale/shift
__global__ void stats_fin_kernel(float* __restrict__ stats,
                                 const float* __restrict__ g,
                                 const float* __restrict__ bt, float Ninv) {
  int j = threadIdx.x;
  if (j >= 64) return;
  float m  = stats[j] * Ninv;
  float v  = stats[64 + j] * Ninv - m * m;
  float sc = g[j] * rsqrtf(v + 1e-5f);
  stats[j]      = sc;
  stats[64 + j] = bt[j] - m * sc;
}

// Encoder GEMM: out[N,128] = relu(bn(h)) @ W2[64,128] + b2   (WMMA)
__global__ __launch_bounds__(128) void enc_gemm_kernel(
    const float* __restrict__ h, const float* __restrict__ ss,
    const unsigned short* __restrict__ w2, const float* __restrict__ b2,
    float* __restrict__ xout, int N) {
  const int wave = threadIdx.x >> 5, lane = threadIdx.x & 31;
  const int g = lane >> 4, lm = lane & 15;
  const int m0 = blockIdx.x * 64 + wave * 16;
  const v16bf* wt = (const v16bf*)w2;   // KT=2, NT=8

  v8f acc[8];
#pragma unroll
  for (int i = 0; i < 8; ++i) acc[i] = v8f_zero();

  const int row = m0 + lm;
#pragma unroll
  for (int kt = 0; kt < 2; ++kt) {
    v16bf a = (row < N) ? load_a_bn(h + row * HID, ss, kt * 32, g) : v16bf_zero();
    v16bf b[8];
#pragma unroll
    for (int nt = 0; nt < 8; ++nt) b[nt] = wt[(kt * 8 + nt) * 32 + lane];
#pragma unroll
    for (int nt = 0; nt < 8; ++nt) acc[nt] = wmma_bf16(a, b[nt], acc[nt]);
  }
#pragma unroll
  for (int nt = 0; nt < 8; ++nt) {
    float bias = b2[nt * 16 + lm];
#pragma unroll
    for (int r = 0; r < 8; ++r) {
      int orow = m0 + r + 8 * g;
      if (orow < N) xout[orow * HID2 + nt * 16 + lm] = acc[nt][r] + bias;
    }
  }
}

// ---------------------------------------------------------------------------
// Edge message + scatter: agg[dst] += relu(x_src[src] + e*We + be)
// 32 threads per edge, float4 per thread; f32 atomics resolve in L2.
// ---------------------------------------------------------------------------
__global__ void edge_scatter_kernel(const float* __restrict__ xs,
                                    const int* __restrict__ src,
                                    const int* __restrict__ dst,
                                    const float* __restrict__ e,
                                    const float* __restrict__ We,
                                    const float* __restrict__ be,
                                    float* __restrict__ agg, int E) {
  int idx = blockIdx.x * blockDim.x + threadIdx.x;
  if (idx >= E * 32) return;
  int edge = idx >> 5;
  int c    = (idx & 31) << 2;
  int s = src[edge], d = dst[edge];
  float ev = e[edge];
  float4 xv = *(const float4*)(xs + (size_t)s * HID2 + c);
  float4 wv = *(const float4*)(We + c);
  float4 bv = *(const float4*)(be + c);
  float m0 = fmaxf(fmaf(ev, wv.x, xv.x + bv.x), 0.f);
  float m1 = fmaxf(fmaf(ev, wv.y, xv.y + bv.y), 0.f);
  float m2 = fmaxf(fmaf(ev, wv.z, xv.z + bv.z), 0.f);
  float m3 = fmaxf(fmaf(ev, wv.w, xv.w + bv.w), 0.f);
  float* p = agg + (size_t)d * HID2 + c;
  unsafeAtomicAdd(p + 0, m0);
  unsafeAtomicAdd(p + 1, m1);
  unsafeAtomicAdd(p + 2, m2);
  unsafeAtomicAdd(p + 3, m3);
}

// ---------------------------------------------------------------------------
// GINE MLP: h2[:,coloff:coloff+64] = relu((x+agg)@W1+b1)@W2+b2   (WMMA x2)
// One block = 4 waves x 16 rows; LDS transposes GEMM1 D-layout -> A-layout.
// B-tiles for each k-step are preloaded so WMMAs issue back-to-back after a
// single s_wait per k-step.
// ---------------------------------------------------------------------------
__global__ __launch_bounds__(128) void conv_mlp_kernel(
    const float* __restrict__ xdst, const float* __restrict__ agg,
    const unsigned short* __restrict__ w1, const float* __restrict__ b1,
    const unsigned short* __restrict__ w2, const float* __restrict__ b2,
    float* __restrict__ h2, int N, int coloff) {
  __shared__ float lds[4][16][65];
  const int wave = threadIdx.x >> 5, lane = threadIdx.x & 31;
  const int g = lane >> 4, lm = lane & 15;
  const int m0 = blockIdx.x * 64 + wave * 16;
  const int row = m0 + lm;

  // GEMM1: [16,128] @ [128,64]
  v8f acc[4];
#pragma unroll
  for (int i = 0; i < 4; ++i) acc[i] = v8f_zero();
  const v16bf* w1t = (const v16bf*)w1;   // KT=4, NT=4
#pragma unroll
  for (int kt = 0; kt < 4; ++kt) {
    v16bf a = (row < N)
                  ? load_a_sum(xdst + (size_t)row * HID2, agg + (size_t)row * HID2,
                               kt * 32, g)
                  : v16bf_zero();
    v16bf b[4];
#pragma unroll
    for (int nt = 0; nt < 4; ++nt) b[nt] = w1t[(kt * 4 + nt) * 32 + lane];
#pragma unroll
    for (int nt = 0; nt < 4; ++nt) acc[nt] = wmma_bf16(a, b[nt], acc[nt]);
  }
  // bias + relu -> LDS (transpose D layout to row-major)
#pragma unroll
  for (int nt = 0; nt < 4; ++nt) {
    float bias = b1[nt * 16 + lm];
#pragma unroll
    for (int r = 0; r < 8; ++r) {
      float v = fmaxf(acc[nt][r] + bias, 0.f);
      lds[wave][r + 8 * g][nt * 16 + lm] = v;
    }
  }
  __syncthreads();

  // GEMM2: [16,64] @ [64,64]
  v8f acc2[4];
#pragma unroll
  for (int i = 0; i < 4; ++i) acc2[i] = v8f_zero();
  const v16bf* w2t = (const v16bf*)w2;   // KT=2, NT=4
#pragma unroll
  for (int kt = 0; kt < 2; ++kt) {
    v16bf a = load_a_lds(&lds[wave][lm][0], kt * 32, g);
    v16bf b[4];
#pragma unroll
    for (int nt = 0; nt < 4; ++nt) b[nt] = w2t[(kt * 4 + nt) * 32 + lane];
#pragma unroll
    for (int nt = 0; nt < 4; ++nt) acc2[nt] = wmma_bf16(a, b[nt], acc2[nt]);
  }
#pragma unroll
  for (int nt = 0; nt < 4; ++nt) {
    float bias = b2[nt * 16 + lm];
#pragma unroll
    for (int r = 0; r < 8; ++r) {
      int orow = m0 + r + 8 * g;
      if (orow < N) h2[(size_t)orow * HID2 + coloff + nt * 16 + lm] = acc2[nt][r] + bias;
    }
  }
}

// x = (relu(h2) + x) * 0.5
__global__ void update_kernel(float* __restrict__ x, const float* __restrict__ h2,
                              int count) {
  int i = blockIdx.x * blockDim.x + threadIdx.x;
  if (i < count) x[i] = (fmaxf(h2[i], 0.f) + x[i]) * 0.5f;
}

// ---------------------------------------------------------------------------
// Prediction head: out[n, step] = (relu?)( relu(h2@W1+b1) . W2 + b2 )
// ---------------------------------------------------------------------------
__global__ __launch_bounds__(128) void pred_kernel(
    const float* __restrict__ h2, const unsigned short* __restrict__ w1,
    const float* __restrict__ b1, const float* __restrict__ w2,
    const float* __restrict__ b2, float* __restrict__ out, int N, int step,
    int relu_out) {
  __shared__ float lds[4][16][65];
  const int wave = threadIdx.x >> 5, lane = threadIdx.x & 31;
  const int g = lane >> 4, lm = lane & 15;
  const int m0 = blockIdx.x * 64 + wave * 16;
  const int row = m0 + lm;

  v8f acc[4];
#pragma unroll
  for (int i = 0; i < 4; ++i) acc[i] = v8f_zero();
  const v16bf* w1t = (const v16bf*)w1;   // KT=4, NT=4
#pragma unroll
  for (int kt = 0; kt < 4; ++kt) {
    v16bf a = (row < N) ? load_a_one(h2 + (size_t)row * HID2, kt * 32, g)
                        : v16bf_zero();
    v16bf b[4];
#pragma unroll
    for (int nt = 0; nt < 4; ++nt) b[nt] = w1t[(kt * 4 + nt) * 32 + lane];
#pragma unroll
    for (int nt = 0; nt < 4; ++nt) acc[nt] = wmma_bf16(a, b[nt], acc[nt]);
  }
#pragma unroll
  for (int nt = 0; nt < 4; ++nt) {
    float bias = b1[nt * 16 + lm];
#pragma unroll
    for (int r = 0; r < 8; ++r)
      lds[wave][r + 8 * g][nt * 16 + lm] = fmaxf(acc[nt][r] + bias, 0.f);
  }
  __syncthreads();

  // final 64-dot in f32, two lanes per row + shuffle combine
  int rrow = lane >> 1, half = lane & 1;
  float s = 0.f;
#pragma unroll
  for (int i = 0; i < 32; ++i)
    s = fmaf(lds[wave][rrow][half * 32 + i], w2[half * 32 + i], s);
  s += __shfl_xor(s, 1, 32);
  int orow = m0 + rrow;
  if (half == 0 && orow < N) {
    float v = s + b2[0];
    if (relu_out) v = fmaxf(v, 0.f);
    out[(size_t)orow * 2 + step] = v;
  }
}

// ---------------------------------------------------------------------------
// Host orchestration
// ---------------------------------------------------------------------------
static inline int cdiv(int a, int b) { return (a + b - 1) / b; }

extern "C" void kernel_launch(void* const* d_in, const int* in_sizes, int n_in,
                              void* d_out, int out_size, void* d_ws, size_t ws_size,
                              hipStream_t stream) {
  (void)in_sizes; (void)n_in; (void)out_size; (void)ws_size;
  const int NC = 25000, NV = 50000, NO = 1000;
  const int EC[6] = {320000, 320000, 50000, 50000, 25000, 25000}; // cv,vc,vo,ov,co,oc

  // d_in layout (recursive dict-insertion order of setup_inputs()):
  // 0:x_cons 1:x_vals 2:x_obj
  // 3..5:cv_{src,dst,e} 6..8:vc 9..11:vo 12..14:ov 15..17:co 18..20:oc
  // 21..26: enc_cons{W1,b1,g,bt,W2,b2} 27..32: enc_vals 33..38: enc_obj
  // 39+6*k: conv[cv,vc,vo,ov,co,oc]{We,be,W1,b1,W2,b2}
  // 75/79: pred_vals[0/1]{W1,b1,W2,b2}   83/87: pred_cons[0/1]
  const float* F[91];
  const int*   I[91];
  for (int i = 0; i < 91; ++i) { F[i] = (const float*)d_in[i]; I[i] = (const int*)d_in[i]; }
  float* out = (float*)d_out;

  // ---- workspace carve-out (~104 MB, fits in 192MB L2) ----
  size_t off = 0;
  float* wsf = (float*)d_ws;
  auto alloc_f = [&](size_t n) { float* p = wsf + off; off += (n + 63) & ~(size_t)63; return p; };
  float* xc  = alloc_f((size_t)NC * HID2);
  float* xv  = alloc_f((size_t)NV * HID2);
  float* xo  = alloc_f((size_t)NO * HID2);
  float* h2c = alloc_f((size_t)NC * HID2);
  float* h2v = alloc_f((size_t)NV * HID2);
  float* h2o = alloc_f((size_t)NO * HID2);
  float* agg = alloc_f((size_t)NV * HID2);   // also reused as encoder h buffer
  float* stats = alloc_f(128);
  unsigned short* ub = (unsigned short*)(wsf + off);
  size_t uo = 0;
  auto alloc_u = [&](size_t n) { unsigned short* p = ub + uo; uo += (n + 15) & ~(size_t)15; return p; };
  unsigned short* encW2bf[3]; for (int t = 0; t < 3; ++t) encW2bf[t]  = alloc_u(64 * 128);
  unsigned short* cW1bf[6];   for (int k = 0; k < 6; ++k) cW1bf[k]    = alloc_u(128 * 64);
  unsigned short* cW2bf[6];   for (int k = 0; k < 6; ++k) cW2bf[k]    = alloc_u(64 * 64);
  unsigned short* pW1bf[4];   for (int k = 0; k < 4; ++k) pW1bf[k]    = alloc_u(128 * 64);

  auto pack = [&](const float* W, unsigned short* dst, int K, int Ncol) {
    int total = (K / 32) * (Ncol / 16) * 512;
    pack_b_kernel<<<cdiv(total, 256), 256, 0, stream>>>(W, dst, K, Ncol);
  };
  // encoder W2 (K=64,N=128); conv W1 (128,64), W2 (64,64); pred W1 (128,64)
  pack(F[25], encW2bf[0], 64, 128);
  pack(F[31], encW2bf[1], 64, 128);
  pack(F[37], encW2bf[2], 64, 128);
  for (int k = 0; k < 6; ++k) {
    int base = 39 + 6 * k;
    pack(F[base + 2], cW1bf[k], 128, 64);
    pack(F[base + 4], cW2bf[k], 64, 64);
  }
  pack(F[75], pW1bf[0], 128, 64);
  pack(F[79], pW1bf[1], 128, 64);
  pack(F[83], pW1bf[2], 128, 64);
  pack(F[87], pW1bf[3], 128, 64);

  // ---- encoders: cons, vals, obj ----
  {
    const float* xin[3]  = {F[0], F[1], F[2]};
    const int    Nn[3]   = {NC, NV, NO};
    const int    pb[3]   = {21, 27, 33};
    float*       X[3]    = {xc, xv, xo};
    for (int t = 0; t < 3; ++t) {
      int N = Nn[t], b = pb[t];
      zero_kernel<<<1, 128, 0, stream>>>(stats, 128);
      enc_h_kernel<<<cdiv(N * HID, 256), 256, 0, stream>>>(xin[t], F[b], F[b + 1], agg, N);
      enc_stats_kernel<<<64, 256, 0, stream>>>(agg, stats, N);
      stats_fin_kernel<<<1, 64, 0, stream>>>(stats, F[b + 2], F[b + 3], 1.f / (float)N);
      enc_gemm_kernel<<<cdiv(N, 64), 128, 0, stream>>>(agg, stats, encW2bf[t], F[b + 5], X[t], N);
    }
  }

  // conv processing order: cv, ov -> h2v ; vc, oc -> h2c ; vo, co -> h2o
  const int    cid[6]    = {0, 3, 1, 5, 2, 4};               // cv, ov, vc, oc, vo, co
  const int    esrc[6]   = {3, 12, 6, 18, 9, 15};
  const float* srcX[6]   = {xc, xo, xv, xo, xv, xc};
  float*       dstX[6]   = {xv, xv, xc, xc, xo, xo};
  float*       h2of[6]   = {h2v, h2v, h2c, h2c, h2o, h2o};
  const int    coloff[6] = {0, 64, 0, 64, 0, 64};
  const int    Nd[6]     = {NV, NV, NC, NC, NO, NO};

  for (int step = 0; step < 2; ++step) {
    for (int layer = 0; layer < 2; ++layer) {
      for (int c = 0; c < 6; ++c) {
        int id = cid[c], pb = 39 + 6 * id, E = EC[id];
        int ei = esrc[c];
        zero_kernel<<<cdiv(Nd[c] * HID2, 256), 256, 0, stream>>>(agg, Nd[c] * HID2);
        edge_scatter_kernel<<<cdiv(E * 32, 256), 256, 0, stream>>>(
            srcX[c], I[ei], I[ei + 1], F[ei + 2], F[pb], F[pb + 1], agg, E);
        conv_mlp_kernel<<<cdiv(Nd[c], 64), 128, 0, stream>>>(
            dstX[c], agg, cW1bf[id], F[pb + 3], cW2bf[id], F[pb + 5],
            h2of[c], Nd[c], coloff[c]);
      }
      update_kernel<<<cdiv(NV * HID2, 256), 256, 0, stream>>>(xv, h2v, NV * HID2);
      update_kernel<<<cdiv(NC * HID2, 256), 256, 0, stream>>>(xc, h2c, NC * HID2);
      update_kernel<<<cdiv(NO * HID2, 256), 256, 0, stream>>>(xo, h2o, NO * HID2);
    }
    // predictions from the last layer's h2 of this step
    int pvb = 75 + 4 * step, pcb = 83 + 4 * step;
    pred_kernel<<<cdiv(NV, 64), 128, 0, stream>>>(
        h2v, pW1bf[step], F[pvb + 1], F[pvb + 2], F[pvb + 3], out, NV, step, 1);
    pred_kernel<<<cdiv(NC, 64), 128, 0, stream>>>(
        h2c, pW1bf[2 + step], F[pcb + 1], F[pcb + 2], F[pcb + 3],
        out + (size_t)NV * 2, NC, step, 0);
  }
}
